// MixtureOfExperts_37065567764964
// MI455X (gfx1250) — compile-verified, matrix operand order checked by reading, not run on hardware
//
#include <hip/hip_runtime.h>
#include <hip/hip_bf16.h>
#include <math.h>

// ---------------------------------------------------------------------------
// Problem constants (match reference)
// ---------------------------------------------------------------------------
#define DM   1024      // D_MODEL
#define DF   4096      // D_FF
#define NE   8         // NUM_EXPERTS
#define TT   8192      // total tokens B*L
#define TILE_M 64      // tokens per MLP block (4 row-tiles of 16)
#define FC   256       // F-chunk per fused iteration

typedef __attribute__((ext_vector_type(16))) __bf16 v16bf;
typedef __attribute__((ext_vector_type(8)))  float  v8f;

struct FragAB { union { unsigned int u[8]; v16bf v; }; };

__device__ __forceinline__ v8f wmma_bf16(v16bf a, v16bf b, v8f c) {
  return __builtin_amdgcn_wmma_f32_16x16x32_bf16(false, a, false, b,
                                                 (short)0, c, false, false);
}

__device__ __forceinline__ unsigned short f2bf(float f) {
  union { float f; unsigned int u; } cv; cv.f = f;
  unsigned int u = cv.u;
  unsigned int r = (u + 0x7FFFu + ((u >> 16) & 1u)) >> 16;   // RNE
  return (unsigned short)r;
}
__device__ __forceinline__ unsigned int pack_bf16x2(float a, float b) {
  return (unsigned int)f2bf(a) | ((unsigned int)f2bf(b) << 16);
}
__device__ __forceinline__ float gelu_exact(float v) {
  return 0.5f * v * (1.0f + erff(v * 0.70710678118654752f));
}

// ---------------------------------------------------------------------------
// Kernel 0: zero output accumulator + expert counts
// ---------------------------------------------------------------------------
__global__ __launch_bounds__(256)
void moe_init_kernel(float* __restrict__ out, int n, int* __restrict__ counts) {
  int i = blockIdx.x * blockDim.x + threadIdx.x;
  if (i < n) out[i] = 0.0f;
  if (i < NE) counts[i] = 0;
}

// ---------------------------------------------------------------------------
// Kernel 1: tiled transpose + fp32->bf16 convert.
//   dst[e][c][r] (bf16) = src[e][r][c] (f32);  R, C multiples of 32.
// Gives WMMA B-matrix layout with contiguous K-pairs (one b32 load / VGPR).
// ---------------------------------------------------------------------------
__global__ __launch_bounds__(256)
void transpose_bf16_kernel(const float* __restrict__ src,
                           unsigned short* __restrict__ dst, int R, int C) {
  __shared__ float tile[32][33];
  const int e  = blockIdx.z;
  const int c0 = blockIdx.x * 32;
  const int r0 = blockIdx.y * 32;
  const int tx = threadIdx.x & 31;
  const int ty = threadIdx.x >> 5;       // 0..7
  const float* s = src + (size_t)e * R * C;
  unsigned short* d = dst + (size_t)e * R * C;
#pragma unroll
  for (int i = 0; i < 32; i += 8)
    tile[ty + i][tx] = s[(size_t)(r0 + ty + i) * C + (c0 + tx)];
  __syncthreads();
#pragma unroll
  for (int i = 0; i < 32; i += 8)
    d[(size_t)(c0 + ty + i) * R + (r0 + tx)] = f2bf(tile[tx][ty + i]);
}

// ---------------------------------------------------------------------------
// Kernel 2: router. One wave (32 lanes) per token. Renormalized top-2 softmax
// weights == 2-way softmax over the two winning logits (normalizer cancels).
// Scatter token into per-expert lists with atomic position allocation.
// ---------------------------------------------------------------------------
__global__ __launch_bounds__(256)
void router_kernel(const float* __restrict__ x,   // [T][DM]
                   const float* __restrict__ Wr,  // [DM][NE]
                   const float* __restrict__ br,  // [NE]
                   int* __restrict__ counts,      // [NE]
                   int* __restrict__ tokList,     // [NE][T]
                   float* __restrict__ wList) {   // [NE][T]
  const int lane = threadIdx.x & 31;
  const int wave = threadIdx.x >> 5;
  const int t = blockIdx.x * 8 + wave;
  if (t >= TT) return;

  float acc[NE];
#pragma unroll
  for (int e = 0; e < NE; ++e) acc[e] = 0.0f;

  const float* xr = x + (size_t)t * DM;
  for (int i = lane; i < DM; i += 32) {
    float xv = xr[i];
    const float* wr = Wr + (size_t)i * NE;
#pragma unroll
    for (int e = 0; e < NE; ++e) acc[e] += xv * wr[e];
  }
#pragma unroll
  for (int off = 16; off > 0; off >>= 1)
#pragma unroll
    for (int e = 0; e < NE; ++e) acc[e] += __shfl_xor(acc[e], off, 32);

  if (lane == 0) {
#pragma unroll
    for (int e = 0; e < NE; ++e) acc[e] += br[e];
    int i0 = 0;
#pragma unroll
    for (int e = 1; e < NE; ++e) if (acc[e] > acc[i0]) i0 = e;   // lowest idx wins ties
    int i1 = (i0 == 0) ? 1 : 0;
#pragma unroll
    for (int e = 0; e < NE; ++e)
      if (e != i0 && acc[e] > acc[i1]) i1 = e;
    float r  = __expf(acc[i1] - acc[i0]);      // <= 1
    float w0 = 1.0f / (1.0f + r);
    float w1 = r * w0;
    int p0 = atomicAdd(&counts[i0], 1);
    tokList[i0 * TT + p0] = t;  wList[i0 * TT + p0] = w0;
    int p1 = atomicAdd(&counts[i1], 1);
    tokList[i1 * TT + p1] = t;  wList[i1 * TT + p1] = w1;
  }
}

// ---------------------------------------------------------------------------
// Kernel 3: fused grouped-GEMM expert MLP, bf16 WMMA.
//   block = (expert e, token tile of 64). 256 threads = 8 waves.
//   Phase A: h = gelu(x @ W1 + b1) for an FC-wide chunk, into LDS.
//            K-outer loop: A-fragments loaded once, reused by 2 B-tiles.
//   Phase B: acc += h @ W2 (each wave owns 128 output columns).
//   Writeout: out[t] += w * (acc + b2)  (atomic; 2 commutative adds => det.)
// ---------------------------------------------------------------------------
__global__ __launch_bounds__(256)
void moe_mlp_kernel(const float* __restrict__ x,            // [T][DM] f32
                    const unsigned short* __restrict__ W1T, // [E][DF][DM] bf16
                    const unsigned short* __restrict__ W2T, // [E][DM][DF] bf16
                    const float* __restrict__ b1,           // [E][DF]
                    const float* __restrict__ b2,           // [E][DM]
                    const int* __restrict__ counts,
                    const int* __restrict__ tokList,        // [E][T]
                    const float* __restrict__ wList,        // [E][T]
                    float* __restrict__ out) {              // [T][DM]
  __shared__ unsigned int x_lds[TILE_M][DM / 2];   // 128 KB, bf16 pairs
  __shared__ unsigned int h_lds[TILE_M][FC / 2];   // 32 KB, bf16 pairs
  __shared__ int   tok_s[TILE_M];
  __shared__ float w_s[TILE_M];

  const int e    = blockIdx.y;
  const int tile = blockIdx.x;
  const int cnt  = counts[e];
  if (tile * TILE_M >= cnt) return;

  const int tid  = threadIdx.x;
  const int lane = tid & 31;
  const int wave = tid >> 5;
  const int row  = lane & 15;            // M/N index within a 16-tile
  const int kb   = (lane >> 4) * 8;      // K base for upper lane half

  if (tid < TILE_M) {
    int r = tile * TILE_M + tid;
    tok_s[tid] = (r < cnt) ? tokList[e * TT + r] : -1;
    w_s[tid]   = (r < cnt) ? wList[e * TT + r] : 0.0f;
  }
  __syncthreads();

  // Gather 64 token rows of x into LDS as packed bf16 (coalesced float2 loads).
  for (int p = tid; p < TILE_M * (DM / 2); p += 256) {
    int r  = p >> 9;            // / (DM/2)
    int cp = p & (DM / 2 - 1);
    int tok = tok_s[r];
    unsigned int packed = 0u;
    if (tok >= 0) {
      float2 f = *(const float2*)(x + (size_t)tok * DM + 2 * cp);
      packed = pack_bf16x2(f.x, f.y);
    }
    x_lds[r][cp] = packed;
  }
  __syncthreads();

  const v8f vzero = {0.f, 0.f, 0.f, 0.f, 0.f, 0.f, 0.f, 0.f};
  v8f acc[4][8];
#pragma unroll
  for (int i = 0; i < 4; ++i)
#pragma unroll
    for (int j = 0; j < 8; ++j) acc[i][j] = vzero;

  const int colbase = wave * 128;        // this wave's output columns (GEMM2)
  const size_t w1e = (size_t)e * DF * DM;
  const size_t w2e = (size_t)e * DM * DF;

  for (int f0 = 0; f0 < DF; f0 += FC) {
    // ---- Phase A: h chunk = gelu(x @ W1 + b1), 64 x FC ----
    // wave covers 32 f-columns: n-tiles (wave*2, wave*2+1), all 4 m-tiles.
    v8f c[2][4];
#pragma unroll
    for (int s = 0; s < 2; ++s)
#pragma unroll
      for (int mt = 0; mt < 4; ++mt) c[s][mt] = vzero;

    const int fcolw = f0 + wave * 32;
    for (int k0 = 0; k0 < DM; k0 += 32) {
      FragAB a[4];
#pragma unroll
      for (int v = 0; v < 8; ++v) {
        int k = k0 + ((v < 4) ? (kb + 2 * v) : (kb + 16 + 2 * (v - 4)));
#pragma unroll
        for (int mt = 0; mt < 4; ++mt)
          a[mt].u[v] = x_lds[mt * 16 + row][k >> 1];
      }
#pragma unroll
      for (int s = 0; s < 2; ++s) {
        const unsigned int* bsrc =
            (const unsigned int*)(W1T + w1e + (size_t)(fcolw + s * 16 + row) * DM);
        FragAB b;
#pragma unroll
        for (int v = 0; v < 8; ++v) {
          int k = k0 + ((v < 4) ? (kb + 2 * v) : (kb + 16 + 2 * (v - 4)));
          b.u[v] = bsrc[k >> 1];
        }
#pragma unroll
        for (int mt = 0; mt < 4; ++mt)
          c[s][mt] = wmma_bf16(a[mt].v, b.v, c[s][mt]);
      }
    }
    // bias + gelu + store to h_lds
    unsigned short* hptr = (unsigned short*)h_lds;
#pragma unroll
    for (int s = 0; s < 2; ++s) {
      const int ncol = wave * 32 + s * 16 + row;       // column within chunk
      float bias = b1[e * DF + f0 + ncol];
#pragma unroll
      for (int mt = 0; mt < 4; ++mt)
#pragma unroll
        for (int v = 0; v < 8; ++v) {
          int m = mt * 16 + v + ((lane >> 4) << 3);    // C-tile VGPR layout
          hptr[m * FC + ncol] = f2bf(gelu_exact(c[s][mt][v] + bias));
        }
    }
    __syncthreads();

    // ---- Phase B: acc += h_chunk @ W2[f0:f0+FC, colbase:+128] ----
    for (int k0 = 0; k0 < FC; k0 += 32) {
      FragAB a[4];
#pragma unroll
      for (int v = 0; v < 8; ++v) {
        int k = k0 + ((v < 4) ? (kb + 2 * v) : (kb + 16 + 2 * (v - 4)));
#pragma unroll
        for (int mt = 0; mt < 4; ++mt)
          a[mt].u[v] = h_lds[mt * 16 + row][k >> 1];
      }
#pragma unroll
      for (int nt = 0; nt < 8; ++nt) {
        const int n = colbase + nt * 16 + row;         // output column d
        const unsigned int* bsrc =
            (const unsigned int*)(W2T + w2e + (size_t)n * DF);
        // prefetch next k-panel of this column into cache
        __builtin_prefetch(bsrc + ((f0 + k0 + 32) >> 1), 0, 0);
        FragAB b;
#pragma unroll
        for (int v = 0; v < 8; ++v) {
          int k = f0 + k0 + ((v < 4) ? (kb + 2 * v) : (kb + 16 + 2 * (v - 4)));
          b.u[v] = bsrc[k >> 1];
        }
#pragma unroll
        for (int mt = 0; mt < 4; ++mt)
          acc[mt][nt] = wmma_bf16(a[mt].v, b.v, acc[mt][nt]);
      }
    }
    __syncthreads();
  }

  // ---- Writeout: out[tok] += w * (acc + b2) ----
#pragma unroll
  for (int mt = 0; mt < 4; ++mt)
#pragma unroll
    for (int nt = 0; nt < 8; ++nt)
#pragma unroll
      for (int v = 0; v < 8; ++v) {
        int m = mt * 16 + v + ((lane >> 4) << 3);
        int tok = tok_s[m];
        if (tok < 0) continue;
        int d = colbase + nt * 16 + row;
        float val = acc[mt][nt][v] + b2[e * DM + d];
        atomicAdd(&out[(size_t)tok * DM + d], w_s[m] * val);
      }
}

// ---------------------------------------------------------------------------
// Launch
// ---------------------------------------------------------------------------
extern "C" void kernel_launch(void* const* d_in, const int* in_sizes, int n_in,
                              void* d_out, int out_size, void* d_ws, size_t ws_size,
                              hipStream_t stream) {
  const float* x  = (const float*)d_in[0];   // [T][DM]
  const float* Wr = (const float*)d_in[1];   // [DM][NE]
  const float* br = (const float*)d_in[2];   // [NE]
  const float* W1 = (const float*)d_in[3];   // [E][DM][DF]
  const float* b1 = (const float*)d_in[4];   // [E][DF]
  const float* W2 = (const float*)d_in[5];   // [E][DF][DM]
  const float* b2 = (const float*)d_in[6];   // [E][DM]
  float* out = (float*)d_out;

  // Workspace layout
  char* ws = (char*)d_ws;
  unsigned short* W1T = (unsigned short*)ws;                    // [E][DF][DM] bf16
  unsigned short* W2T = W1T + (size_t)NE * DF * DM;             // [E][DM][DF] bf16
  int*   counts  = (int*)(W2T + (size_t)NE * DM * DF);
  int*   tokList = counts + 16;                                 // [E][T]
  float* wList   = (float*)(tokList + NE * TT);                 // [E][T]

  const int nOut = TT * DM;
  moe_init_kernel<<<(nOut + 255) / 256, 256, 0, stream>>>(out, nOut, counts);

  // W1: [E][DM][DF] -> W1T [E][DF][DM]
  transpose_bf16_kernel<<<dim3(DF / 32, DM / 32, NE), 256, 0, stream>>>(W1, W1T, DM, DF);
  // W2: [E][DF][DM] -> W2T [E][DM][DF]
  transpose_bf16_kernel<<<dim3(DM / 32, DF / 32, NE), 256, 0, stream>>>(W2, W2T, DF, DM);

  router_kernel<<<TT / 8, 256, 0, stream>>>(x, Wr, br, counts, tokList, wList);

  moe_mlp_kernel<<<dim3(TT / TILE_M, NE), 256, 0, stream>>>(
      x, W1T, W2T, b1, b2, counts, tokList, wList, out);
}
